// ClaHi_GAT_37958920962326
// MI455X (gfx1250) — matrix operand order repaired and checked by reading, not compiled, not run on hardware
//
#include <hip/hip_runtime.h>

// ---------------------------------------------------------------------------
// CDNA5 / gfx1250 implementation.
//  - GEMMs use v_wmma_f32_16x16x32_f16 (f16 in, f32 accumulate).
//  - Big GEMMs: 32x32 register tile per wave (2x2 wmma, 16 FLOP/B from L2).
//  - Serial LSTM step GEMM: 16x16 tile per wave (max wave parallelism for the
//    latency-bound recurrent chain).
//  - A: row-major f16, K padded to 32, rows padded to 32.
//  - Bt: pre-transposed (N x K) row-major f16, rows padded to 32 -> each
//    lane's B fragment is one contiguous 32B load; A fragment also one contiguous 32B load
//    (k = 16*half + s in both A and B, not the ISA table order).
// ---------------------------------------------------------------------------

typedef __attribute__((ext_vector_type(16))) _Float16 v16h;
typedef __attribute__((ext_vector_type(8)))  _Float16 v8h;
typedef __attribute__((ext_vector_type(8)))  float    v8f;

#define NV   346   // nodes / batch
#define NP   352   // NV padded (multiple of 32)
#define LL   50    // sequence length
#define DP   320   // 300 padded to 32
#define HH   300   // hidden
#define G4   1200  // 4*H
#define G4P  1216  // 1200 padded to 32 (Bt rows)
#define K1P  608   // 600 padded to 32
#define FC2P 32    // FC2t rows (N=1 padded)

// ---------------------------- reductions -----------------------------------
__device__ __forceinline__ float blockReduceSum(float v) {
    __shared__ float red[256];
    int t = threadIdx.x;
    red[t] = v; __syncthreads();
    for (int s = 128; s > 0; s >>= 1) { if (t < s) red[t] += red[t + s]; __syncthreads(); }
    float r = red[0]; __syncthreads();
    return r;
}
__device__ __forceinline__ float blockReduceMax(float v) {
    __shared__ float red[256];
    int t = threadIdx.x;
    red[t] = v; __syncthreads();
    for (int s = 128; s > 0; s >>= 1) { if (t < s) red[t] = fmaxf(red[t], red[t + s]); __syncthreads(); }
    float r = red[0]; __syncthreads();
    return r;
}

__device__ __forceinline__ float applyAct(float v, int ACT) {
    if (ACT == 1) return tanhf(v);
    if (ACT == 2) return (v > 0.f) ? v : (expf(v) - 1.f);
    return v;
}

// ---------------------- WMMA GEMM, 32x32 tile per wave ---------------------
// C[M,N] = act( A[M,K] * Bt[N,K]^T + bias[N] + Cadd[M,N] )
// A rows padded to 32, Bt rows padded to 32, K multiple of 32.
template <int ACT>
__global__ __launch_bounds__(256)
void gemm_wmma32(const _Float16* __restrict__ A, int lda,
                 const _Float16* __restrict__ Bt, int ldb,
                 float* __restrict__ C, int ldc,
                 const float* __restrict__ Cadd, int ldca,
                 const float* __restrict__ bias,
                 int M, int N, int K)
{
    const int tN = (N + 31) >> 5;
    const int tM = (M + 31) >> 5;
    const int wave = blockIdx.x * (blockDim.x >> 5) + (threadIdx.x >> 5);
    if (wave >= tM * tN) return;
    const int tm   = wave / tN;
    const int tn   = wave % tN;
    const int lane = threadIdx.x & 31;
    const int half = lane >> 4;
    const int r    = lane & 15;

    const _Float16* a0p = A  + (size_t)(tm * 32 + r) * lda;
    const _Float16* a1p = a0p + (size_t)16 * lda;
    const _Float16* b0p = Bt + (size_t)(tn * 32 + r) * ldb;
    const _Float16* b1p = b0p + (size_t)16 * ldb;

    v8f c00 = {}, c01 = {}, c10 = {}, c11 = {};
    for (int k0 = 0; k0 < K; k0 += 32) {
        union { v16h v; v8h h[2]; } a0, a1, b0, b1;
        a0.v = *(const v16h*)(a0p + k0 + 16 * half);
        a1.v = *(const v16h*)(a1p + k0 + 16 * half);
        b0.v = *(const v16h*)(b0p + k0 + 16 * half);
        b1.v = *(const v16h*)(b1p + k0 + 16 * half);
        c00 = __builtin_amdgcn_wmma_f32_16x16x32_f16(false, a0.v, false, b0.v, (short)0, c00, false, false);
        c01 = __builtin_amdgcn_wmma_f32_16x16x32_f16(false, a0.v, false, b1.v, (short)0, c01, false, false);
        c10 = __builtin_amdgcn_wmma_f32_16x16x32_f16(false, a1.v, false, b0.v, (short)0, c10, false, false);
        c11 = __builtin_amdgcn_wmma_f32_16x16x32_f16(false, a1.v, false, b1.v, (short)0, c11, false, false);
    }

    const int mb = tm * 32, nb = tn * 32;
#pragma unroll
    for (int st = 0; st < 4; ++st) {
        v8f* accp = (st == 0) ? &c00 : (st == 1) ? &c01 : (st == 2) ? &c10 : &c11;
        int mbase = mb + ((st >> 1) << 4);
        int nbase = nb + ((st & 1) << 4);
#pragma unroll
        for (int vr = 0; vr < 8; ++vr) {
            int m = mbase + vr + 8 * half;
            int n = nbase + r;
            if (m < M && n < N) {
                float v = (*accp)[vr];
                if (bias) v += bias[n];
                if (Cadd) v += Cadd[(size_t)m * ldca + n];
                C[(size_t)m * ldc + n] = applyAct(v, ACT);
            }
        }
    }
}

// ---------------------- WMMA GEMM, 16x16 tile per wave ---------------------
// Used for the serial LSTM recurrent step (latency-bound, wants max waves).
__global__ __launch_bounds__(256)
void gemm_wmma16(const _Float16* __restrict__ A, int lda,
                 const _Float16* __restrict__ Bt, int ldb,
                 float* __restrict__ C, int ldc,
                 const float* __restrict__ Cadd, int ldca,
                 int M, int N, int K)
{
    const int tilesN = (N + 15) >> 4;
    const int tilesM = (M + 15) >> 4;
    const int wave   = blockIdx.x * (blockDim.x >> 5) + (threadIdx.x >> 5);
    if (wave >= tilesM * tilesN) return;
    const int tm   = wave / tilesN;
    const int tn   = wave % tilesN;
    const int lane = threadIdx.x & 31;
    const int half = lane >> 4;
    const int r    = lane & 15;

    const _Float16* arow = A  + (size_t)(tm * 16 + r) * lda;
    const _Float16* brow = Bt + (size_t)(tn * 16 + r) * ldb;

    v8f acc = {};
    for (int k0 = 0; k0 < K; k0 += 32) {
        union { v16h v; v8h h[2]; } a, b;
        a.v = *(const v16h*)(arow + k0 + 16 * half);
        b.v = *(const v16h*)(brow + k0 + 16 * half);
        acc = __builtin_amdgcn_wmma_f32_16x16x32_f16(false, a.v, false, b.v,
                                                     (short)0, acc, false, false);
    }
#pragma unroll
    for (int vr = 0; vr < 8; ++vr) {
        int m = tm * 16 + vr + 8 * half;
        int n = tn * 16 + r;
        if (m < M && n < N) {
            float v = acc[vr];
            if (Cadd) v += Cadd[(size_t)m * ldca + n];
            C[(size_t)m * ldc + n] = v;
        }
    }
}

// ---------------------------- staging kernels ------------------------------
__global__ void k_zero_h(_Float16* p, size_t n) {
    size_t i = blockIdx.x * (size_t)blockDim.x + threadIdx.x;
    if (i < n) p[i] = (_Float16)0.f;
}

// dst[Rp,Cp] f16 = src[R,C] f32, zero padded
__global__ void k_cvt_f16(const float* __restrict__ src, int R, int C,
                          _Float16* __restrict__ dst, int Rp, int Cp)
{
    size_t id = blockIdx.x * (size_t)blockDim.x + threadIdx.x;
    size_t tot = (size_t)Rp * Cp;
    if (id >= tot) return;
    int c = (int)(id % Cp);
    int r = (int)(id / Cp);
    float v = (r < R && c < C) ? src[(size_t)r * C + c] : 0.f;
    dst[id] = (_Float16)v;
}

// dst[Cp,Rp] f16 = transpose(src[R,C] f32), zero padded. LDS-tiled.
__global__ void k_cvt_f16_T(const float* __restrict__ src, int R, int C,
                            _Float16* __restrict__ dst, int Rp, int Cp)
{
    __shared__ float t[32][33];
    int cb = blockIdx.x * 32;   // src col block == dst row block
    int rb = blockIdx.y * 32;   // src row block == dst col block
    int tx = threadIdx.x, ty = threadIdx.y;
    for (int i = ty; i < 32; i += 8) {
        int r = rb + i, c = cb + tx;
        t[i][tx] = (r < R && c < C) ? src[(size_t)r * C + c] : 0.f;
    }
    __syncthreads();
    for (int i = ty; i < 32; i += 8) {
        int dr = cb + i;   // dst row (src col)
        int dc = rb + tx;  // dst col (src row)
        if (dr < Cp && dc < Rp) dst[(size_t)dr * Rp + dc] = (_Float16)t[tx][i];
    }
}

// x[l][n][d] = emb[tok[n][l]][d]  (f16, padded to NP x DP)
__global__ void k_embed(const int* __restrict__ tok, const float* __restrict__ emb,
                        _Float16* __restrict__ x, int D)
{
    size_t id = blockIdx.x * (size_t)blockDim.x + threadIdx.x;
    size_t tot = (size_t)LL * NP * DP;
    if (id >= tot) return;
    int d = (int)(id % DP);
    size_t t2 = id / DP;
    int n = (int)(t2 % NP);
    int l = (int)(t2 / NP);
    float v = 0.f;
    if (n < NV && d < D) {
        int tk = tok[(size_t)n * LL + l];
        v = emb[(size_t)tk * D + d];
    }
    x[id] = (_Float16)v;
}

// LSTM pointwise: G[n, 0:4H] pre-activations -> (c, h)
__global__ void k_lstm_cell(const float* __restrict__ G, float* __restrict__ c,
                            _Float16* __restrict__ h16, int h16ld,
                            _Float16* __restrict__ seq, int seqld,
                            float* __restrict__ hT, int hTld,
                            int first)
{
    int id = blockIdx.x * blockDim.x + threadIdx.x;
    if (id >= NV * HH) return;
    int n = id / HH, j = id % HH;
    const float* g = G + (size_t)n * G4;
    float gi = g[j], gf = g[HH + j], gg = g[2 * HH + j], go = g[3 * HH + j];
    float cp = first ? 0.f : c[id];
    float si = 1.f / (1.f + expf(-gi));
    float sf = 1.f / (1.f + expf(-gf));
    float so = 1.f / (1.f + expf(-go));
    float cn = sf * cp + si * tanhf(gg);
    float hn = so * tanhf(cn);
    c[id] = cn;
    h16[(size_t)n * h16ld + j] = (_Float16)hn;
    if (seq) seq[(size_t)n * seqld + j] = (_Float16)hn;
    if (hT)  hT [(size_t)n * hTld  + j] = hn;
}

// scalar = dot(hs[0], U)
__global__ void k_dot1(const float* __restrict__ hs, const float* __restrict__ U,
                       int K, float* __restrict__ out)
{
    float s = 0.f;
    for (int j = threadIdx.x; j < K; j += blockDim.x) s += hs[j] * U[j];
    s = blockReduceSum(s);
    if (threadIdx.x == 0) out[0] = s;
}

// g[i] = sigmoid(dot(hs[i], W) + hcU), i = 1..NV-1 (block i-1)
__global__ void k_gate_g(const float* __restrict__ hs, const float* __restrict__ W,
                         const float* __restrict__ hcU, float* __restrict__ g, int K)
{
    int i = blockIdx.x + 1;
    const float* row = hs + (size_t)i * K;
    float s = 0.f;
    for (int j = threadIdx.x; j < K; j += blockDim.x) s += row[j] * W[j];
    s = blockReduceSum(s);
    if (threadIdx.x == 0) g[i] = 1.f / (1.f + expf(-(s + hcU[0])));
}

// colsum[j] = sum_{i=1..NV-1} hs[i][j]
__global__ void k_colsum(const float* __restrict__ hs, float* __restrict__ cs, int K)
{
    int j = blockIdx.x * blockDim.x + threadIdx.x;
    if (j >= K) return;
    float s = 0.f;
    for (int i = 1; i < NV; ++i) s += hs[(size_t)i * K + j];
    cs[j] = s;
}

// Ag[NP, 2*1200] f16 = [hs | gate(hs)]
__global__ void k_build_Ag(const float* __restrict__ hs, const float* __restrict__ g,
                           const float* __restrict__ cs, _Float16* __restrict__ Ag)
{
    size_t id = blockIdx.x * (size_t)blockDim.x + threadIdx.x;
    size_t tot = (size_t)NP * 2 * G4;
    if (id >= tot) return;
    int j2 = (int)(id % (2 * G4));
    int i  = (int)(id / (2 * G4));
    float v = 0.f;
    if (i < NV) {
        if (j2 < G4) v = hs[(size_t)i * G4 + j2];
        else {
            int j = j2 - G4;
            if (i == 0) v = hs[j];
            else {
                float gi = g[i];
                v = gi * cs[j] + (1.f - gi) * ((float)(NV - 1) * hs[j]);
            }
        }
    }
    Ag[id] = (_Float16)v;
}

// s1[i] = dot(Wh[i], a[0:F]); s2[i] = dot(Wh[i], a[F:2F])
__global__ void k_s12(const float* __restrict__ Wh, const float* __restrict__ a,
                      float* __restrict__ s1, float* __restrict__ s2, int F)
{
    int i = blockIdx.x;
    const float* row = Wh + (size_t)i * F;
    float x1 = 0.f, x2 = 0.f;
    for (int j = threadIdx.x; j < F; j += blockDim.x) {
        float w = row[j];
        x1 += w * a[j];
        x2 += w * a[F + j];
    }
    x1 = blockReduceSum(x1);
    x2 = blockReduceSum(x2);
    if (threadIdx.x == 0) { s1[i] = x1; s2[i] = x2; }
}

// att row i: softmax_j( mask ? leaky(s1[i]+s2[j]) : -9e15 ), stored f16 padded
__global__ void k_att(const float* __restrict__ s1, const float* __restrict__ s2,
                      const int* __restrict__ adj, _Float16* __restrict__ att)
{
    int i = blockIdx.x;
    float s1i = s1[i];
    float mx = -3.0e38f;
    for (int j = threadIdx.x; j < NV; j += blockDim.x) {
        float e = s1i + s2[j];
        e = (e > 0.f) ? e : 0.01f * e;
        e = (adj[(size_t)i * NV + j] > 0) ? e : -9.0e15f;
        mx = fmaxf(mx, e);
    }
    mx = blockReduceMax(mx);
    float sm = 0.f;
    for (int j = threadIdx.x; j < NV; j += blockDim.x) {
        float e = s1i + s2[j];
        e = (e > 0.f) ? e : 0.01f * e;
        e = (adj[(size_t)i * NV + j] > 0) ? e : -9.0e15f;
        float p = expf(e - mx);
        sm += p;
        att[(size_t)i * NP + j] = (_Float16)p;
    }
    sm = blockReduceSum(sm);
    float inv = 1.f / sm;
    for (int j = threadIdx.x; j < NV; j += blockDim.x)
        att[(size_t)i * NP + j] = (_Float16)((float)att[(size_t)i * NP + j] * inv);
    for (int j = NV + threadIdx.x; j < NP; j += blockDim.x)
        att[(size_t)i * NP + j] = (_Float16)0.f;
}

// cat4[NP,4800] f16 = [hc, hs2, hc*hs2, hc-hs2], hc = hs2[0]
__global__ void k_build_cat4(const float* __restrict__ hs2, _Float16* __restrict__ cat)
{
    size_t id = blockIdx.x * (size_t)blockDim.x + threadIdx.x;
    size_t tot = (size_t)NP * 4800;
    if (id >= tot) return;
    int j4 = (int)(id % 4800);
    int i  = (int)(id / 4800);
    float v = 0.f;
    if (i < NV) {
        int b = j4 / G4, j = j4 % G4;
        float hc = hs2[j];
        float hi = hs2[(size_t)i * G4 + j];
        v = (b == 0) ? hc : (b == 1) ? hi : (b == 2) ? hc * hi : (hc - hi);
    }
    cat[id] = (_Float16)v;
}

// ea[0:1200] = mean_i hs2[i][:], ea[1200:2400] = sum_i batt[i]*hs2[i][:]
__global__ void k_ea(const float* __restrict__ hs2, const float* __restrict__ batt,
                     float* __restrict__ ea)
{
    int j = blockIdx.x * blockDim.x + threadIdx.x;
    if (j >= 2 * G4) return;
    if (j < G4) {
        float s = 0.f;
        for (int i = 0; i < NV; ++i) s += hs2[(size_t)i * G4 + j];
        ea[j] = s / (float)NV;
    } else {
        int jj = j - G4;
        float s = 0.f;
        for (int i = 0; i < NV; ++i) s += batt[i] * hs2[(size_t)i * G4 + jj];
        ea[j] = s;
    }
}

// out = softmax(ea @ Wlin.T + blin) over 2 classes
__global__ void k_head(const float* __restrict__ ea, const float* __restrict__ Wlin,
                       const float* __restrict__ blin, float* __restrict__ out, int K)
{
    float s0 = 0.f, s1 = 0.f;
    for (int j = threadIdx.x; j < K; j += blockDim.x) {
        float e = ea[j];
        s0 += e * Wlin[j];
        s1 += e * Wlin[K + j];
    }
    s0 = blockReduceSum(s0);
    s1 = blockReduceSum(s1);
    if (threadIdx.x == 0) {
        float l0 = s0 + blin[0], l1 = s1 + blin[1];
        float m = fmaxf(l0, l1);
        float e0 = expf(l0 - m), e1 = expf(l1 - m);
        out[0] = e0 / (e0 + e1);
        out[1] = e1 / (e0 + e1);
    }
}

// ---------------------------------------------------------------------------
extern "C" void kernel_launch(void* const* d_in, const int* in_sizes, int n_in,
                              void* d_out, int out_size, void* d_ws, size_t ws_size,
                              hipStream_t stream)
{
    (void)in_sizes; (void)n_in; (void)out_size; (void)ws_size;

    const int*   tokens  = (const int*)  d_in[0];
    const int*   adj     = (const int*)  d_in[1];
    const float* emb     = (const float*)d_in[2];
    const float* Wih_l0f = (const float*)d_in[3];
    const float* Whh_l0f = (const float*)d_in[4];
    const float* b_l0f   = (const float*)d_in[5];
    const float* Wih_l0b = (const float*)d_in[6];
    const float* Whh_l0b = (const float*)d_in[7];
    const float* b_l0b   = (const float*)d_in[8];
    const float* Wih_l1f = (const float*)d_in[9];
    const float* Whh_l1f = (const float*)d_in[10];
    const float* b_l1f   = (const float*)d_in[11];
    const float* Wih_l1b = (const float*)d_in[12];
    const float* Whh_l1b = (const float*)d_in[13];
    const float* b_l1b   = (const float*)d_in[14];
    const float* W_gat   = (const float*)d_in[15];
    const float* a_gat   = (const float*)d_in[16];
    const float* gate_W  = (const float*)d_in[17];
    const float* gate_U  = (const float*)d_in[18];
    const float* FC1     = (const float*)d_in[19];
    const float* FC2     = (const float*)d_in[20];
    const float* Wlin    = (const float*)d_in[21];
    const float* blin    = (const float*)d_in[22];
    float* out = (float*)d_out;

    // ---- workspace bump allocator (256B aligned) ----
    char* base = (char*)d_ws;
    size_t off = 0;
    auto alloc = [&](size_t bytes) -> void* {
        off = (off + 255) & ~(size_t)255;
        void* p = base + off;
        off += bytes;
        return p;
    };
    _Float16* x16    = (_Float16*)alloc((size_t)LL * NP * DP  * 2);
    _Float16* o016   = (_Float16*)alloc((size_t)LL * NP * K1P * 2);
    float*    xproj  = (float*)   alloc((size_t)LL * NP * G4  * 4);
    float*    Gb     = (float*)   alloc((size_t)NV * G4 * 4);
    float*    cb     = (float*)   alloc((size_t)NV * HH * 4);
    _Float16* h16    = (_Float16*)alloc((size_t)NP * DP * 2);
    _Float16* w_ih0f = (_Float16*)alloc((size_t)G4P * DP * 2);
    _Float16* w_hh0f = (_Float16*)alloc((size_t)G4P * DP * 2);
    _Float16* w_ih0b = (_Float16*)alloc((size_t)G4P * DP * 2);
    _Float16* w_hh0b = (_Float16*)alloc((size_t)G4P * DP * 2);
    _Float16* w_ih1f = (_Float16*)alloc((size_t)G4P * K1P * 2);
    _Float16* w_hh1f = (_Float16*)alloc((size_t)G4P * DP * 2);
    _Float16* w_ih1b = (_Float16*)alloc((size_t)G4P * K1P * 2);
    _Float16* w_hh1b = (_Float16*)alloc((size_t)G4P * DP * 2);
    _Float16* wgatT  = (_Float16*)alloc((size_t)G4P * 2400 * 2);
    _Float16* fc1T   = (_Float16*)alloc((size_t)9600 * 4800 * 2);
    _Float16* fc2T   = (_Float16*)alloc((size_t)FC2P * 9600 * 2);
    float*    hsA    = (float*)   alloc((size_t)NV * G4 * 4);
    float*    hsB    = (float*)   alloc((size_t)NV * G4 * 4);
    float*    hcU    = (float*)   alloc(256);
    float*    gvec   = (float*)   alloc((size_t)NP * 4);
    float*    colsum = (float*)   alloc((size_t)G4 * 4);
    _Float16* Ag16   = (_Float16*)alloc((size_t)NP * 2 * G4 * 2);
    float*    Wh     = (float*)   alloc((size_t)NV * G4 * 4);
    _Float16* WhT16  = (_Float16*)alloc((size_t)G4P * NP * 2);
    float*    s1v    = (float*)   alloc((size_t)NP * 4);
    float*    s2v    = (float*)   alloc((size_t)NP * 4);
    _Float16* att16  = (_Float16*)alloc((size_t)NP * NP * 2);
    _Float16* cat416 = (_Float16*)alloc((size_t)NP * 4800 * 2);
    float*    hc3    = (float*)   alloc((size_t)NV * 9600 * 4);
    _Float16* hc316  = (_Float16*)alloc((size_t)NP * 9600 * 2);
    float*    battv  = (float*)   alloc((size_t)NP * 4);
    float*    eab    = (float*)   alloc((size_t)2 * G4 * 4);

    auto g1 = [](size_t n) { return dim3((unsigned)((n + 255) / 256)); };

    // 32x32-tile GEMM for throughput-bound layers
    auto gemm32 = [&](int act, const _Float16* A, int lda, const _Float16* Bt, int ldb,
                      float* C, int ldc, const float* Cadd, int ldca, const float* bias,
                      int M, int Nn, int K) {
        int tiles = ((M + 31) / 32) * ((Nn + 31) / 32);
        dim3 grid((tiles + 7) / 8);
        if (act == 0)      gemm_wmma32<0><<<grid, 256, 0, stream>>>(A, lda, Bt, ldb, C, ldc, Cadd, ldca, bias, M, Nn, K);
        else if (act == 1) gemm_wmma32<1><<<grid, 256, 0, stream>>>(A, lda, Bt, ldb, C, ldc, Cadd, ldca, bias, M, Nn, K);
        else               gemm_wmma32<2><<<grid, 256, 0, stream>>>(A, lda, Bt, ldb, C, ldc, Cadd, ldca, bias, M, Nn, K);
    };
    // 16x16-tile GEMM for the latency-bound LSTM step
    auto gemm16 = [&](const _Float16* A, int lda, const _Float16* Bt, int ldb,
                      float* C, int ldc, const float* Cadd, int ldca,
                      int M, int Nn, int K) {
        int tiles = ((M + 15) / 16) * ((Nn + 15) / 16);
        dim3 grid((tiles + 7) / 8);
        gemm_wmma16<<<grid, 256, 0, stream>>>(A, lda, Bt, ldb, C, ldc, Cadd, ldca, M, Nn, K);
    };

    // ---- 1. weight staging to f16 (Bt = Npad x Kpad layout, rows padded to 32)
    k_cvt_f16<<<g1((size_t)G4P * DP), 256, 0, stream>>>(Wih_l0f, G4, 300, w_ih0f, G4P, DP);
    k_cvt_f16<<<g1((size_t)G4P * DP), 256, 0, stream>>>(Whh_l0f, G4, 300, w_hh0f, G4P, DP);
    k_cvt_f16<<<g1((size_t)G4P * DP), 256, 0, stream>>>(Wih_l0b, G4, 300, w_ih0b, G4P, DP);
    k_cvt_f16<<<g1((size_t)G4P * DP), 256, 0, stream>>>(Whh_l0b, G4, 300, w_hh0b, G4P, DP);
    k_cvt_f16<<<g1((size_t)G4P * K1P), 256, 0, stream>>>(Wih_l1f, G4, 600, w_ih1f, G4P, K1P);
    k_cvt_f16<<<g1((size_t)G4P * DP), 256, 0, stream>>>(Whh_l1f, G4, 300, w_hh1f, G4P, DP);
    k_cvt_f16<<<g1((size_t)G4P * K1P), 256, 0, stream>>>(Wih_l1b, G4, 600, w_ih1b, G4P, K1P);
    k_cvt_f16<<<g1((size_t)G4P * DP), 256, 0, stream>>>(Whh_l1b, G4, 300, w_hh1b, G4P, DP);
    {
        dim3 tb(32, 8);
        dim3 gg((G4P + 31) / 32, (2400 + 31) / 32);     // W_gat (2400,1200) -> (1216,2400)
        k_cvt_f16_T<<<gg, tb, 0, stream>>>(W_gat, 2400, G4, wgatT, 2400, G4P);
        dim3 gf((9600 + 31) / 32, (4800 + 31) / 32);    // FC1 (4800,9600) -> (9600,4800)
        k_cvt_f16_T<<<gf, tb, 0, stream>>>(FC1, 4800, 9600, fc1T, 4800, 9600);
        dim3 g2((FC2P + 31) / 32, (9600 + 31) / 32);    // FC2 (9600,1) -> (32,9600)
        k_cvt_f16_T<<<g2, tb, 0, stream>>>(FC2, 9600, 1, fc2T, 9600, FC2P);
    }

    // ---- 2. embedding gather -> x16 (L, NP, DP) f16 ----
    k_embed<<<g1((size_t)LL * NP * DP), 256, 0, stream>>>(tokens, emb, x16, 300);
    k_zero_h<<<g1((size_t)LL * NP * K1P), 256, 0, stream>>>(o016, (size_t)LL * NP * K1P);

    // ---- 3. LSTM scans (x@Wih.T + b batched outside the serial loop) ----
    auto scan = [&](const _Float16* Ax, int Kp, const _Float16* wih, const _Float16* whh,
                    const float* bias, int rev, int seqcol, float* hTp) {
        gemm32(0, Ax, Kp, wih, Kp, xproj, G4, nullptr, 0, bias, LL * NP, G4, Kp);
        k_zero_h<<<g1((size_t)NP * DP), 256, 0, stream>>>(h16, (size_t)NP * DP);
        for (int it = 0; it < LL; ++it) {
            int t = rev ? (LL - 1 - it) : it;
            gemm16(h16, DP, whh, DP, Gb, G4,
                   xproj + (size_t)t * NP * G4, G4, NV, G4, DP);
            _Float16* seq = (seqcol >= 0) ? (o016 + (size_t)t * NP * K1P + seqcol) : nullptr;
            k_lstm_cell<<<g1((size_t)NV * HH), 256, 0, stream>>>(
                Gb, cb, h16, DP, seq, K1P,
                (it == LL - 1) ? hTp : nullptr, G4, (it == 0) ? 1 : 0);
        }
    };
    scan(x16,  DP,  w_ih0f, w_hh0f, b_l0f, 0, 0,   hsA + 0);    // l0 fwd
    scan(x16,  DP,  w_ih0b, w_hh0b, b_l0b, 1, 300, hsA + 300);  // l0 bwd
    scan(o016, K1P, w_ih1f, w_hh1f, b_l1f, 0, -1,  hsA + 600);  // l1 fwd
    scan(o016, K1P, w_ih1b, w_hh1b, b_l1b, 1, -1,  hsA + 900);  // l1 bwd

    // ---- 4. two GAT+gate rounds ----
    auto gat_round = [&](const float* src, float* dst) {
        k_dot1<<<1, 256, 0, stream>>>(src, gate_U, G4, hcU);
        k_gate_g<<<NV - 1, 256, 0, stream>>>(src, gate_W, hcU, gvec, G4);
        k_colsum<<<g1((size_t)G4), 256, 0, stream>>>(src, colsum, G4);
        k_build_Ag<<<g1((size_t)NP * 2 * G4), 256, 0, stream>>>(src, gvec, colsum, Ag16);
        gemm32(0, Ag16, 2 * G4, wgatT, 2 * G4, Wh, G4, nullptr, 0, nullptr, NV, G4, 2 * G4);
        k_s12<<<NV, 256, 0, stream>>>(Wh, a_gat, s1v, s2v, G4);
        k_zero_h<<<g1((size_t)NP * NP), 256, 0, stream>>>(att16, (size_t)NP * NP);
        k_att<<<NV, 256, 0, stream>>>(s1v, s2v, adj, att16);
        dim3 tb(32, 8), tg((G4P + 31) / 32, (NP + 31) / 32);    // Wh (346,1200) -> (1216,352)
        k_cvt_f16_T<<<tg, tb, 0, stream>>>(Wh, NV, G4, WhT16, NP, G4P);
        gemm32(2, att16, NP, WhT16, NP, dst, G4, nullptr, 0, nullptr, NV, G4, NP); // elu fused
    };
    gat_round(hsA, hsB);  // hs0 -> hs1
    gat_round(hsB, hsA);  // hs1 -> hs2 (in hsA)

    // ---- 5. head ----
    k_build_cat4<<<g1((size_t)NP * 4800), 256, 0, stream>>>(hsA, cat416);
    gemm32(1, cat416, 4800, fc1T, 4800, hc3, 9600, nullptr, 0, nullptr, NV, 9600, 4800); // tanh
    k_cvt_f16<<<g1((size_t)NP * 9600), 256, 0, stream>>>(hc3, NV, 9600, hc316, NP, 9600);
    gemm32(1, hc316, 9600, fc2T, 9600, battv, 1, nullptr, 0, nullptr, NV, 1, 9600);      // tanh
    k_ea<<<g1((size_t)2 * G4), 256, 0, stream>>>(hsA, battv, eab);
    k_head<<<1, 256, 0, stream>>>(eab, Wlin, blin, out, 2 * G4);
}
